// VMM_1812476199093
// MI455X (gfx1250) — compile-verified, matrix operand order checked by reading, not run on hardware
//
#include <hip/hip_runtime.h>

#define B_   2
#define H_   64
#define Wd_  64
#define C_   96
#define DIN_ 192
#define NST_ 16
#define RNK_ 6
#define K_   4
#define L_   4096
#define ND_  38   // RNK + 2*NST

typedef __attribute__((ext_vector_type(16))) _Float16 v16h;
typedef __attribute__((ext_vector_type(8)))  float    v8f;

// Load a 16-lane-row fragment (A-style; B uses same mapping with lane = N).
// Per ISA 7.12.2 (16-bit A 16x32): lanes 0-15 hold K = k0+{0..7} in v0-3 and
// k0+{16..23} in v4-7; lanes 16-31 hold K = k0+{8..15} / k0+{24..31}.
__device__ __forceinline__ v16h frag_ld(const float* __restrict__ p, int ld,
                                        int row, int k0, int half) {
  const float* r0 = p + (size_t)row * ld + k0 + 8 * half;
  v16h f;
#pragma unroll
  for (int i = 0; i < 8; ++i) {
    f[i]     = (_Float16)r0[i];
    f[i + 8] = (_Float16)r0[i + 16];
  }
  return f;
}

// ---- CDNA5 async global->LDS copy (ASYNCcnt path, CDNA5 ISA) --------
// LDS byte address = low 32 bits of the flat LDS-aperture pointer (§10.2).
__device__ __forceinline__ void async_ld_b32(void* lds_dst, const void* gsrc) {
  unsigned l = (unsigned)(unsigned long long)lds_dst;
  unsigned long long g = (unsigned long long)gsrc;
  asm volatile("global_load_async_to_lds_b32 %0, %1, off"
               :: "v"(l), "v"(g) : "memory");
}
__device__ __forceinline__ void wait_async0() {
  asm volatile("s_wait_asynccnt 0x0" ::: "memory");
}

// ---------------- in_proj: xz = x(8192x96) @ W_in^T(96x384) ----------------
// split: cols [0,192) -> xct (b,d,l) channels-first for conv; [192,384) -> z (b,l,d)
__global__ void vmm_inproj(const float* __restrict__ x, const float* __restrict__ Win,
                           float* __restrict__ xct, float* __restrict__ z) {
  int wid  = blockIdx.x * (blockDim.x >> 5) + (threadIdx.x >> 5); // 0..3071
  int lane = threadIdx.x & 31;
  int half = lane >> 4, cl = lane & 15;
  int mt = wid / 6, ng = wid % 6;            // 512 M-tiles x 6 groups of 64 cols
  int mrow = mt * 16 + cl;
  v8f acc[4] = {};
#pragma unroll
  for (int ks = 0; ks < 3; ++ks) {
    int k0 = ks * 32;
    __builtin_prefetch(x + (size_t)mrow * C_ + ((k0 + 32) % C_), 0, 1);
    v16h a = frag_ld(x, C_, mrow, k0, half);
#pragma unroll
    for (int nt = 0; nt < 4; ++nt) {
      int nrow = ng * 64 + nt * 16 + cl;
      v16h bf = frag_ld(Win, C_, nrow, k0, half);
      acc[nt] = __builtin_amdgcn_wmma_f32_16x16x32_f16(
          false, a, false, bf, (short)0, acc[nt], false, false);
    }
  }
#pragma unroll
  for (int nt = 0; nt < 4; ++nt) {
    int col = ng * 64 + nt * 16 + cl;
#pragma unroll
    for (int r = 0; r < 8; ++r) {
      int row = mt * 16 + r + 8 * half;      // global row = b*L + l
      int b = row >> 12, l = row & (L_ - 1);
      float v = acc[nt][r];
      if (col < DIN_) xct[((size_t)(b * DIN_ + col)) * L_ + l] = v;
      else            z[(size_t)row * DIN_ + (col - DIN_)]     = v;
    }
  }
}

// ------------- depthwise 3x3 SAME conv + bias + SiLU -> xconv (b,d,l) -------
__global__ void vmm_dwconv(const float* __restrict__ xct, const float* __restrict__ cw,
                           const float* __restrict__ cb, float* __restrict__ xconv) {
  int idx = blockIdx.x * blockDim.x + threadIdx.x;   // covers B*DIN*L exactly
  int l  = idx & (L_ - 1);
  int bd = idx >> 12;
  int d  = bd % DIN_;
  int h = l >> 6, w = l & 63;
  const float* base = xct + (size_t)bd * L_;
  float s = cb[d];
#pragma unroll
  for (int i = 0; i < 3; ++i) {
    int hh = h + i - 1;
    if ((unsigned)hh >= (unsigned)H_) continue;
#pragma unroll
    for (int j = 0; j < 3; ++j) {
      int ww = w + j - 1;
      if ((unsigned)ww >= (unsigned)Wd_) continue;
      s += base[hh * 64 + ww] * cw[d * 9 + i * 3 + j];
    }
  }
  xconv[idx] = s / (1.f + __expf(-s));  // SiLU
}

// ---- x_proj as WMMA GEMM: xdbl[b,k,r,l] = sum_d xpw[k,r,d]*xconv[b,d,scan[k,l]]
// M=38 (3 tiles of 16, rows >=38 clamped+discarded), N=4096, K=192.
__global__ void vmm_xproj(const float* __restrict__ xconv, const float* __restrict__ xpw,
                          const int* __restrict__ scan_ids, float* __restrict__ xdbl) {
  int wid  = blockIdx.x * (blockDim.x >> 5) + (threadIdx.x >> 5); // 0..2047
  int lane = threadIdx.x & 31;
  int half = lane >> 4, cl = lane & 15;
  int bk = wid >> 8;                    // 0..7
  int nt = wid & 255;                   // 16-col tile of l
  int b = bk >> 2, k = bk & 3;
  int l0 = nt * 16;
  int sid = scan_ids[k * L_ + l0 + cl]; // gathered spatial index for this lane's column
  const float* xcb = xconv + (size_t)b * DIN_ * L_;
  const float* wk  = xpw + (size_t)k * ND_ * DIN_;
  v8f acc[3] = {};
#pragma unroll
  for (int ks = 0; ks < 6; ++ks) {
    int k0 = ks * 32;
    // B fragment: column l of gathered XS, kmap over channel d (stride-L gather)
    v16h bf;
#pragma unroll
    for (int i = 0; i < 8; ++i) {
      int d0 = k0 + 8 * half + i;
      bf[i]     = (_Float16)xcb[(size_t)d0 * L_ + sid];
      bf[i + 8] = (_Float16)xcb[(size_t)(d0 + 16) * L_ + sid];
    }
#pragma unroll
    for (int mt = 0; mt < 3; ++mt) {
      int r = mt * 16 + cl;
      int rr = r < ND_ ? r : 0;         // clamp pad rows (results discarded)
      v16h af = frag_ld(wk, DIN_, rr, k0, half);
      acc[mt] = __builtin_amdgcn_wmma_f32_16x16x32_f16(
          false, af, false, bf, (short)0, acc[mt], false, false);
    }
  }
#pragma unroll
  for (int mt = 0; mt < 3; ++mt)
#pragma unroll
    for (int r = 0; r < 8; ++r) {
      int row = mt * 16 + r + 8 * half;
      if (row < ND_)
        xdbl[((size_t)(b * K_ + k) * ND_ + row) * L_ + l0 + cl] = acc[mt][r];
    }
}

// -------- fused selective scan: delta inline, h-recurrence, y, pooled -------
// LDS staging via CDNA5 async global->LDS loads (ASYNCcnt + s_wait_asynccnt).
__global__ void vmm_scan(const float* __restrict__ xconv, const float* __restrict__ xdbl,
                         const int* __restrict__ scan_ids, const float* __restrict__ dt_w,
                         const float* __restrict__ dt_b, const float* __restrict__ A_logs,
                         const float* __restrict__ Ds, float* __restrict__ ys,
                         float* __restrict__ pooled) {
  __shared__ float s_xd[ND_ * 64];          // rows: 0..5 dts, 6..21 B, 22..37 C
  __shared__ float s_xs[DIN_ * 65];         // stride-65 pad: bank-conflict free
  __shared__ int   s_sid[64];
  int b = blockIdx.x >> 2, k = blockIdx.x & 3;
  int tid = threadIdx.x;                    // d channel, 192 threads
  int kd = k * DIN_ + tid;
  float Ad[NST_];
#pragma unroll
  for (int n = 0; n < NST_; ++n) Ad[n] = -__expf(A_logs[kd * NST_ + n]);
  float dtw[RNK_];
#pragma unroll
  for (int r = 0; r < RNK_; ++r) dtw[r] = dt_w[kd * RNK_ + r];
  float dtb = dt_b[kd], Dd = Ds[kd];
  float hst[NST_] = {};
  float ssum = 0.f;
  size_t xdbase = (size_t)(b * K_ + k) * ND_ * L_;
  size_t xcb = (size_t)(b * DIN_ + tid) * L_;
  size_t ysb = (size_t)((b * K_ + k) * DIN_ + tid) * L_;
  for (int l0 = 0; l0 < L_; l0 += 64) {
    __syncthreads();                        // previous chunk fully consumed
    if (tid < 64) s_sid[tid] = scan_ids[k * L_ + l0 + tid];
    for (int o = tid; o < ND_ * 64; o += DIN_)
      async_ld_b32(&s_xd[o], &xdbl[xdbase + (size_t)(o >> 6) * L_ + l0 + (o & 63)]);
    __syncthreads();                        // s_sid visible (asyncs still in flight)
    for (int li = 0; li < 64; ++li)
      async_ld_b32(&s_xs[tid * 65 + li], &xconv[xcb + s_sid[li]]);
    wait_async0();                          // drain this wave's ASYNCcnt
    __syncthreads();                        // publish async LDS writes block-wide
    for (int li = 0; li < 64; ++li) {
      float dtsum = dtb;
#pragma unroll
      for (int r = 0; r < RNK_; ++r) dtsum += dtw[r] * s_xd[r * 64 + li];
      float delta = dtsum > 20.f ? dtsum : log1pf(__expf(dtsum));  // softplus
      float xv  = s_xs[tid * 65 + li];
      float dbx = delta * xv;
      float y   = Dd * xv;
#pragma unroll
      for (int n = 0; n < NST_; ++n) {
        float dA = __expf(delta * Ad[n]);
        float Bn = s_xd[(RNK_ + n) * 64 + li];
        float Cn = s_xd[(RNK_ + NST_ + n) * 64 + li];
        hst[n] = dA * hst[n] + dbx * Bn;
        y += hst[n] * Cn;
      }
      ys[ysb + l0 + li] = y;
      ssum += y;
    }
  }
  pooled[(size_t)(b * DIN_ + tid) * K_ + k] = ssum * (1.f / L_);
}

// ------------------------- shuffle-attn gate (tiny) -------------------------
__global__ void vmm_gate(const float* __restrict__ pooled, const float* __restrict__ gw,
                         const float* __restrict__ gb, float* __restrict__ gate) {
  int t = blockIdx.x * blockDim.x + threadIdx.x;
  if (t >= B_ * DIN_) return;
  int b = t / DIN_, d = t % DIN_;
  float p[K_];
#pragma unroll
  for (int i = 0; i < K_; ++i) p[i] = pooled[(size_t)(b * DIN_ + d) * K_ + i];
#pragma unroll
  for (int o = 0; o < K_; ++o) {
    float a = gb[d * K_ + o];
#pragma unroll
    for (int i = 0; i < K_; ++i) a += p[i] * gw[(d * K_ + o) * K_ + i];
    gate[(size_t)(b * DIN_ + d) * K_ + o] = 1.f / (1.f + __expf(-a));
  }
}

// ---- combine: inverse scatter + gate + sum_k + LayerNorm + silu(z) gate ----
__global__ void vmm_combine(const float* __restrict__ ys, const float* __restrict__ gate,
                            const int* __restrict__ inv_ids, const float* __restrict__ z,
                            const float* __restrict__ ln_g, const float* __restrict__ ln_b,
                            float* __restrict__ y2) {
  __shared__ float s_red[DIN_];
  __shared__ float s_stat[2];
  int bl = blockIdx.x;                       // b*L + l
  int b = bl >> 12, l = bl & (L_ - 1);
  int d = threadIdx.x;
  float v = 0.f;
#pragma unroll
  for (int k = 0; k < K_; ++k) {
    int src = inv_ids[k * L_ + l];
    v += gate[(size_t)(b * DIN_ + d) * K_ + k] *
         ys[(size_t)((b * K_ + k) * DIN_ + d) * L_ + src];
  }
  s_red[d] = v;
  __syncthreads();
  for (int s = 96; s >= 6; s >>= 1) { if (d < s) s_red[d] += s_red[d + s]; __syncthreads(); }
  if (d == 0) { float t = 0.f; for (int i = 0; i < 6; ++i) t += s_red[i]; s_stat[0] = t * (1.f / DIN_); }
  __syncthreads();
  float mu = s_stat[0];
  float dv = v - mu;
  s_red[d] = dv * dv;
  __syncthreads();
  for (int s = 96; s >= 6; s >>= 1) { if (d < s) s_red[d] += s_red[d + s]; __syncthreads(); }
  if (d == 0) { float t = 0.f; for (int i = 0; i < 6; ++i) t += s_red[i]; s_stat[1] = t * (1.f / DIN_); }
  __syncthreads();
  float rstd = rsqrtf(s_stat[1] + 1e-5f);
  float zy = z[(size_t)bl * DIN_ + d];
  float sz = zy / (1.f + __expf(-zy));
  y2[(size_t)bl * DIN_ + d] = (dv * rstd * ln_g[d] + ln_b[d]) * sz;
}

// --------------- out_proj: out = y2(8192x192) @ W_out^T(192x96) -------------
__global__ void vmm_outproj(const float* __restrict__ y2, const float* __restrict__ Wout,
                            float* __restrict__ out) {
  int wid  = blockIdx.x * (blockDim.x >> 5) + (threadIdx.x >> 5); // mt 0..511
  int lane = threadIdx.x & 31;
  int half = lane >> 4, cl = lane & 15;
  int mrow = wid * 16 + cl;
  v8f acc[6] = {};
#pragma unroll
  for (int ks = 0; ks < 6; ++ks) {
    int k0 = ks * 32;
    __builtin_prefetch(y2 + (size_t)mrow * DIN_ + ((k0 + 32) % DIN_), 0, 1);
    v16h a = frag_ld(y2, DIN_, mrow, k0, half);
#pragma unroll
    for (int nt = 0; nt < 6; ++nt) {
      v16h bf = frag_ld(Wout, DIN_, nt * 16 + cl, k0, half);
      acc[nt] = __builtin_amdgcn_wmma_f32_16x16x32_f16(
          false, a, false, bf, (short)0, acc[nt], false, false);
    }
  }
#pragma unroll
  for (int nt = 0; nt < 6; ++nt)
#pragma unroll
    for (int r = 0; r < 8; ++r) {
      int row = wid * 16 + r + 8 * half;
      out[(size_t)row * C_ + nt * 16 + cl] = acc[nt][r];
    }
}

extern "C" void kernel_launch(void* const* d_in, const int* in_sizes, int n_in,
                              void* d_out, int out_size, void* d_ws, size_t ws_size,
                              hipStream_t stream) {
  (void)in_sizes; (void)n_in; (void)out_size; (void)ws_size;
  const float* x      = (const float*)d_in[0];
  const float* W_in   = (const float*)d_in[1];
  const float* conv_w = (const float*)d_in[2];
  const float* conv_b = (const float*)d_in[3];
  const float* xpw    = (const float*)d_in[4];
  const float* dt_w   = (const float*)d_in[5];
  const float* dt_b   = (const float*)d_in[6];
  const float* A_logs = (const float*)d_in[7];
  const float* Ds     = (const float*)d_in[8];
  const float* gate_w = (const float*)d_in[9];
  const float* gate_b = (const float*)d_in[10];
  const float* ln_g   = (const float*)d_in[11];
  const float* ln_b   = (const float*)d_in[12];
  const float* W_out  = (const float*)d_in[13];
  const int* scan_ids = (const int*)d_in[14];
  const int* inv_ids  = (const int*)d_in[15];
  float* out = (float*)d_out;

  float* ws = (float*)d_ws;
  size_t off = 0;
  float* xct    = ws + off; off += (size_t)B_ * DIN_ * L_;      // 6.3 MB
  float* z      = ws + off; off += (size_t)B_ * L_ * DIN_;      // 6.3 MB
  float* xconv  = ws + off; off += (size_t)B_ * DIN_ * L_;      // 6.3 MB
  float* xdbl   = ws + off; off += (size_t)B_ * K_ * ND_ * L_;  // 5.0 MB
  float* ys     = ws + off; off += (size_t)B_ * K_ * DIN_ * L_; // 25 MB
  float* pooled = ws + off; off += (size_t)B_ * DIN_ * K_;
  float* gate   = ws + off; off += (size_t)B_ * DIN_ * K_;
  float* y2     = ws + off; off += (size_t)B_ * L_ * DIN_;      // 6.3 MB

  vmm_inproj <<<384, 256, 0, stream>>>(x, W_in, xct, z);
  vmm_dwconv <<<6144, 256, 0, stream>>>(xct, conv_w, conv_b, xconv);
  vmm_xproj  <<<256, 256, 0, stream>>>(xconv, xpw, scan_ids, xdbl);
  vmm_scan   <<<8, 192, 0, stream>>>(xconv, xdbl, scan_ids, dt_w, dt_b,
                                     A_logs, Ds, ys, pooled);
  vmm_gate   <<<3, 128, 0, stream>>>(pooled, gate_w, gate_b, gate);
  vmm_combine<<<8192, 192, 0, stream>>>(ys, gate, inv_ids, z, ln_g, ln_b, y2);
  vmm_outproj<<<64, 256, 0, stream>>>(y2, W_out, out);
}
